// LowLevelGRU_simplified_14207751815707
// MI455X (gfx1250) — matrix operand order repaired and checked by reading, not compiled, not run on hardware
//
#include <hip/hip_runtime.h>
#include <hip/hip_bf16.h>

// Problem dims (compile-time constants from the reference)
#define IN_DIM  1024
#define H_DIM   2048
#define Y_DIM   1024
#define ROLLOUT 512

typedef __attribute__((ext_vector_type(16))) __bf16 v16bf;
typedef __attribute__((ext_vector_type(8)))  __bf16 v8bf;
typedef __attribute__((ext_vector_type(8)))  float  v8f;

// ---------- helpers ----------
__device__ __forceinline__ unsigned short f2bf(float f) {
    unsigned u = __float_as_uint(f);
    unsigned r = u + 0x7FFFu + ((u >> 16) & 1u);   // round-to-nearest-even
    return (unsigned short)(r >> 16);
}
__device__ __forceinline__ float bf2f(unsigned short u) {
    return __uint_as_float(((unsigned)u) << 16);
}

// ---------- fp32 -> bf16 elementwise ----------
__global__ void cvt_bf16_kernel(const float* __restrict__ src,
                                unsigned short* __restrict__ dst, int n) {
    int i = blockIdx.x * blockDim.x + threadIdx.x;
    if (i < n) dst[i] = f2bf(src[i]);
}

// ---------- fp32 -> bf16 transposed convert ----------
// dst[c*R + r] = bf16(src[(row_off + r)*C + c]); launch block(32,8), grid(R/32, C/32)
__global__ void transpose_cvt_kernel(const float* __restrict__ src,
                                     unsigned short* __restrict__ dst,
                                     int R, int C, int row_off) {
    __shared__ float tile[32][33];
    int rb = blockIdx.x * 32, cb = blockIdx.y * 32;
    for (int i = threadIdx.y; i < 32; i += 8)
        tile[i][threadIdx.x] = src[(size_t)(row_off + rb + i) * C + cb + threadIdx.x];
    __syncthreads();
    for (int i = threadIdx.y; i < 32; i += 8)
        dst[(size_t)(cb + i) * R + rb + threadIdx.x] = f2bf(tile[threadIdx.x][i]);
}

// ---------- bf16 WMMA GEMM:  C[M,N] = A[M,K] @ B^T  (+ bias[N]) ----------
// A bf16 row-major [M,K]; BT bf16 [N,K] (i.e. B transposed); C fp32 [M,N].
// grid = (N/64, M/64), block = 128 (4 waves). Wave w: rows [by*64+16w, +16),
// 64 columns -> 4 accumulators with A-fragment reuse.
// Fragment layouts used here (A and B share contiguous k order):
//   A 16x32: lane m=l15, halves hold K 0..15 / 16..31 -> 2x b128
//   B 32x16: lane n=l15, halves hold K 0..15 / 16..31 -> contiguous 32B from BT
//   C 16x16 f32: VGPR i -> row i + 8*half, col l15
__global__ void wmma_gemm_bf16(const unsigned short* __restrict__ A,
                               const unsigned short* __restrict__ BT,
                               const float* __restrict__ bias,
                               float* __restrict__ C,
                               int M, int N, int K) {
    const int lane = threadIdx.x & 31;
    const int wv   = threadIdx.x >> 5;
    const int half = lane >> 4;        // 0: lanes 0-15, 1: lanes 16-31
    const int l15  = lane & 15;
    const int mbase = blockIdx.y * 64 + wv * 16;
    const int nbase = blockIdx.x * 64;

    v8f acc[4];
    #pragma unroll
    for (int j = 0; j < 4; ++j) acc[j] = (v8f){0.f,0.f,0.f,0.f,0.f,0.f,0.f,0.f};

    const unsigned short* arow = A + (size_t)(mbase + l15) * K;  // lane's A row
    for (int k0 = 0; k0 < K; k0 += 32) {
        // A fragment: two contiguous 8-element (16B) runs
        const unsigned short* pa = arow + k0 + half * 16;
        v8bf alo = *(const v8bf*)pa;          // K offs +0..7
        v8bf ahi = *(const v8bf*)(pa + 8);    // K offs +8..15
        v16bf a = __builtin_shufflevector(alo, ahi,
                    0,1,2,3,4,5,6,7,8,9,10,11,12,13,14,15);
        #pragma unroll
        for (int j = 0; j < 4; ++j) {
            // B fragment: one contiguous 16-element (32B) run from BT
            int n = nbase + j * 16 + l15;
            v16bf b = *(const v16bf*)(BT + (size_t)n * K + k0 + half * 16);
            acc[j] = __builtin_amdgcn_wmma_f32_16x16x32_bf16(
                false, a, false, b, (short)0, acc[j], false, false);
        }
    }
    // store C (+bias)
    #pragma unroll
    for (int j = 0; j < 4; ++j) {
        int n = nbase + j * 16 + l15;
        float bv = bias ? bias[n] : 0.f;
        #pragma unroll
        for (int i = 0; i < 8; ++i)
            C[(size_t)(mbase + i + half * 8) * N + n] = acc[j][i] + bv;
    }
}

// ---------- one recurrent step ----------
// One wave per output column c (2048 waves). Weights transposed bf16:
// WuhT/WchT[c*H_DIM + r] contiguous -> coalesced packed b64 loads.
// gxu/gxc rows already contain x-projection + bias.
__global__ void gru_step_kernel(const float* __restrict__ hprev,
                                float* __restrict__ hnew,
                                const float* __restrict__ gxu,
                                const float* __restrict__ gxc,
                                const unsigned short* __restrict__ WuhT,
                                const unsigned short* __restrict__ WchT,
                                unsigned short* __restrict__ hbf_row) {
    __shared__ float sh[H_DIM];
    for (int i = threadIdx.x; i < H_DIM; i += blockDim.x) sh[i] = hprev[i];
    __syncthreads();

    const int lane = threadIdx.x & 31;
    const int c = blockIdx.x * (blockDim.x >> 5) + (threadIdx.x >> 5);
    const unsigned short* wu = WuhT + (size_t)c * H_DIM;
    const unsigned short* wc = WchT + (size_t)c * H_DIM;

    float au = 0.f, ac = 0.f;
    for (int r = lane * 4; r < H_DIM; r += 128) {       // 4 bf16 per lane per iter
        uint2 pu = *(const uint2*)(wu + r);
        uint2 pc = *(const uint2*)(wc + r);
        float h0 = sh[r], h1 = sh[r + 1], h2 = sh[r + 2], h3 = sh[r + 3];
        au = fmaf(bf2f((unsigned short)(pu.x & 0xFFFFu)), h0, au);
        au = fmaf(bf2f((unsigned short)(pu.x >> 16)),     h1, au);
        au = fmaf(bf2f((unsigned short)(pu.y & 0xFFFFu)), h2, au);
        au = fmaf(bf2f((unsigned short)(pu.y >> 16)),     h3, au);
        ac = fmaf(bf2f((unsigned short)(pc.x & 0xFFFFu)), h0, ac);
        ac = fmaf(bf2f((unsigned short)(pc.x >> 16)),     h1, ac);
        ac = fmaf(bf2f((unsigned short)(pc.y & 0xFFFFu)), h2, ac);
        ac = fmaf(bf2f((unsigned short)(pc.y >> 16)),     h3, ac);
    }
    // wave32 reduction
    #pragma unroll
    for (int off = 16; off > 0; off >>= 1) {
        au += __shfl_down(au, off, 32);
        ac += __shfl_down(ac, off, 32);
    }
    if (lane == 0) {
        float gu = au + gxu[c];
        float gc = ac + gxc[c];
        float Lu = 1.f / (1.f + __expf(-gu));
        float ct = tanhf(gc);
        float hp = sh[c];
        float hn = Lu * ct + (1.f - Lu) * hp;
        hnew[c]    = hn;
        hbf_row[c] = f2bf(hn);
    }
}

extern "C" void kernel_launch(void* const* d_in, const int* in_sizes, int n_in,
                              void* d_out, int out_size, void* d_ws, size_t ws_size,
                              hipStream_t stream) {
    const float* x    = (const float*)d_in[0];   // [1, T, IN_DIM]
    const float* h0   = (const float*)d_in[1];   // [H_DIM]
    const float* Wc   = (const float*)d_in[2];   // [IN+H, H]
    const float* Wu   = (const float*)d_in[3];   // [IN+H, H]
    const float* bc   = (const float*)d_in[4];   // [H]
    const float* bu   = (const float*)d_in[5];   // [H]
    const float* Why  = (const float*)d_in[6];   // [H, Y]
    const float* by   = (const float*)d_in[7];   // [Y]
    float* out        = (float*)d_out;           // ys [T,Y] ++ h_final [H]

    // workspace carve-up (bytes, 256-aligned)
    char* ws = (char*)d_ws;
    size_t off = 0;
    auto carve = [&](size_t bytes) { char* p = ws + off; off = (off + bytes + 255) & ~(size_t)255; return p; };
    unsigned short* Xbf   = (unsigned short*)carve((size_t)ROLLOUT * IN_DIM * 2);
    unsigned short* WuxT  = (unsigned short*)carve((size_t)IN_DIM * H_DIM * 2);  // [H_DIM, IN_DIM]
    unsigned short* WcxT  = (unsigned short*)carve((size_t)IN_DIM * H_DIM * 2);  // [H_DIM, IN_DIM]
    unsigned short* WuhT  = (unsigned short*)carve((size_t)H_DIM * H_DIM * 2);   // [H_DIM, H_DIM]
    unsigned short* WchT  = (unsigned short*)carve((size_t)H_DIM * H_DIM * 2);   // [H_DIM, H_DIM]
    unsigned short* WhyT  = (unsigned short*)carve((size_t)H_DIM * Y_DIM * 2);   // [Y_DIM, H_DIM]
    float*          Gxu   = (float*)carve((size_t)ROLLOUT * H_DIM * 4);
    float*          Gxc   = (float*)carve((size_t)ROLLOUT * H_DIM * 4);
    float*          hhist = (float*)carve((size_t)(ROLLOUT + 1) * H_DIM * 4);
    unsigned short* Hbf   = (unsigned short*)carve((size_t)ROLLOUT * H_DIM * 2);
    (void)ws_size; (void)in_sizes; (void)n_in; (void)out_size;

    // 1) conversions: X -> bf16; all weight blocks -> transposed bf16 [N,K]
    {
        int n = ROLLOUT * IN_DIM;
        cvt_bf16_kernel<<<(n + 255) / 256, 256, 0, stream>>>(x, Xbf, n);
        dim3 tb(32, 8);
        // x-blocks of Wu/Wc: src rows 0..IN_DIM-1, cols H_DIM -> [H_DIM, IN_DIM]
        dim3 tgx(IN_DIM / 32, H_DIM / 32);
        transpose_cvt_kernel<<<tgx, tb, 0, stream>>>(Wu, WuxT, IN_DIM, H_DIM, 0);
        transpose_cvt_kernel<<<tgx, tb, 0, stream>>>(Wc, WcxT, IN_DIM, H_DIM, 0);
        // h-blocks of Wu/Wc: src rows IN_DIM..IN_DIM+H_DIM-1 -> [H_DIM, H_DIM]
        dim3 tgh(H_DIM / 32, H_DIM / 32);
        transpose_cvt_kernel<<<tgh, tb, 0, stream>>>(Wu, WuhT, H_DIM, H_DIM, IN_DIM);
        transpose_cvt_kernel<<<tgh, tb, 0, stream>>>(Wc, WchT, H_DIM, H_DIM, IN_DIM);
        // Wh_y [H_DIM, Y_DIM] -> [Y_DIM, H_DIM]
        dim3 tgy(H_DIM / 32, Y_DIM / 32);
        transpose_cvt_kernel<<<tgy, tb, 0, stream>>>(Why, WhyT, H_DIM, Y_DIM, 0);
    }

    // 2) hoisted x-projections (WMMA): Gx = X @ Wx + b
    {
        dim3 blk(128);
        dim3 g1(H_DIM / 64, ROLLOUT / 64);
        wmma_gemm_bf16<<<g1, blk, 0, stream>>>(Xbf, WuxT, bu, Gxu, ROLLOUT, H_DIM, IN_DIM);
        wmma_gemm_bf16<<<g1, blk, 0, stream>>>(Xbf, WcxT, bc, Gxc, ROLLOUT, H_DIM, IN_DIM);
    }

    // 3) initial hidden state -> hhist row 0
    hipMemcpyAsync(hhist, h0, H_DIM * sizeof(float), hipMemcpyDeviceToDevice, stream);

    // 4) sequential recurrence: 512 small fused-step kernels (graph-captured)
    {
        const int waves_per_blk = 8;                 // 256 threads
        const int nblk = H_DIM / waves_per_blk;      // 256 blocks = 2048 waves/columns
        for (int t = 0; t < ROLLOUT; ++t) {
            gru_step_kernel<<<nblk, waves_per_blk * 32, 0, stream>>>(
                hhist + (size_t)t * H_DIM,
                hhist + (size_t)(t + 1) * H_DIM,
                Gxu + (size_t)t * H_DIM,
                Gxc + (size_t)t * H_DIM,
                WuhT, WchT,
                Hbf + (size_t)t * H_DIM);
        }
    }

    // 5) deferred output projection (WMMA): ys = H @ Wh_y + by -> d_out
    {
        dim3 blk(128);
        dim3 g3(Y_DIM / 64, ROLLOUT / 64);
        wmma_gemm_bf16<<<g3, blk, 0, stream>>>(Hbf, WhyT, by, out, ROLLOUT, Y_DIM, H_DIM);
    }

    // 6) h_final -> tail of d_out
    hipMemcpyAsync(out + (size_t)ROLLOUT * Y_DIM, hhist + (size_t)ROLLOUT * H_DIM,
                   H_DIM * sizeof(float), hipMemcpyDeviceToDevice, stream);
}
